// TransformerModel_13984413516319
// MI455X (gfx1250) — matrix operand order repaired and checked by reading, not compiled, not run on hardware
//
#include <hip/hip_runtime.h>
#include <hip/hip_bf16.h>
#include <math.h>

// Model dims (fixed by the reference)
#define BB    2
#define SS    2048
#define TT    (BB * SS)          // 4096 tokens
#define HH    1536
#define NHD   24
#define NKVH  4
#define DHD   64
#define KVD   (NKVH * DHD)       // 256
#define MFF   8960
#define LLAY  8

typedef __attribute__((ext_vector_type(16))) __bf16        v16bf;
typedef __attribute__((ext_vector_type(8)))  float         v8f;
typedef __attribute__((ext_vector_type(8)))  unsigned short u16x8;
typedef __attribute__((ext_vector_type(16))) unsigned short u16x16;
typedef __attribute__((ext_vector_type(4)))  int            v4i;

// ---------------- CDNA5 async global->LDS copy (ASYNCcnt) -----------------
#if defined(__has_builtin)
#if __has_builtin(__builtin_amdgcn_global_load_async_to_lds_b128)
#define HAVE_ASYNC_LDS 1
#endif
#endif
#ifndef HAVE_ASYNC_LDS
#define HAVE_ASYNC_LDS 0
#endif

typedef __attribute__((address_space(1))) v4i g1_v4i;   // global <4 x i32>
typedef __attribute__((address_space(3))) v4i l3_v4i;   // LDS    <4 x i32>

__device__ __forceinline__ void cp16_to_lds(const unsigned short* gptr,
                                            unsigned short* lptr) {
#if HAVE_ASYNC_LDS
    __builtin_amdgcn_global_load_async_to_lds_b128((g1_v4i*)gptr,
                                                   (l3_v4i*)lptr, 0, 0);
#else
    *(u16x8*)lptr = *(const u16x8*)gptr;
#endif
}

__device__ __forceinline__ void async_publish_wait() {
#if HAVE_ASYNC_LDS
#if __has_builtin(__builtin_amdgcn_s_wait_asynccnt)
    __builtin_amdgcn_s_wait_asynccnt(0);
#else
    asm volatile("s_wait_asynccnt 0x0" ::: "memory");
#endif
#endif
}

// ---------- bf16 <-> f32 helpers (RNE) ----------
__device__ __forceinline__ unsigned short f2bf(float f) {
    union { float f; unsigned u; } x; x.f = f;
    unsigned r = x.u + 0x7FFFu + ((x.u >> 16) & 1u);
    return (unsigned short)(r >> 16);
}
__device__ __forceinline__ float bf2f(unsigned short h) {
    union { unsigned u; float f; } x; x.u = ((unsigned)h) << 16;
    return x.f;
}

// ---------- fragment loads (A and B both use contiguous k = 16*half + s) ----------
// A-operand (16x32 bf16): two contiguous 8-element (16B) chunks per lane.
__device__ __forceinline__ v16bf frag_split(const unsigned short* p0,
                                            const unsigned short* p1) {
    u16x8 lo = *(const u16x8*)p0;
    u16x8 hi = *(const u16x8*)p1;
    union { u16x16 u; v16bf b; } un;
    un.u = __builtin_shufflevector(lo, hi, 0,1,2,3,4,5,6,7,8,9,10,11,12,13,14,15);
    return un.b;
}
// B-operand (32x16 bf16, weight stored [N,K]): 16 contiguous bf16 per lane.
__device__ __forceinline__ v16bf frag_contig(const unsigned short* p) {
    union { u16x16 u; v16bf b; } un;
    un.u = *(const u16x16*)p;
    return un.b;
}

__device__ __forceinline__ v8f wmma_bf16(v16bf a, v16bf b, v8f c) {
    return __builtin_amdgcn_wmma_f32_16x16x32_bf16(false, a, false, b,
                                                   (short)0, c, false, false);
}

// ============================================================================
// GEMM: C[T,N] = A[T,K](bf16) x Bt[N,K](bf16)^T  (+ bias / epilogue variants)
// Block: 256 thr = 8 waves; wave tile 64x32 (4x2 WMMA tiles); WG tile 128x128.
// A/B 128x32 tiles double-buffered in LDS via async global->LDS copies.
// EPI: 0 = bf16 store (+bias)          1 = fp32 residual +=
//      2 = bf16 store sigmoid(aux)*acc 3 = bf16 store gelu(acc+bias)
// ============================================================================
template <int EPI>
__global__ __launch_bounds__(256)
void gemm_bf16(const unsigned short* __restrict__ A,
               const unsigned short* __restrict__ Bt,
               const float* __restrict__ bias,
               const unsigned short* __restrict__ aux,
               float* __restrict__ outF,
               unsigned short* __restrict__ outB,
               int T, int N, int K)
{
    __shared__ __align__(32) unsigned short As[2][128 * 32];
    __shared__ __align__(32) unsigned short Bs[2][128 * 32];

    const int tid  = threadIdx.x;
    const int lane = tid & 31;
    const int w    = tid >> 5;
    const int wm   = w >> 2;         // 0..1
    const int wn   = w & 3;          // 0..3
    const int half = lane >> 4;
    const int l15  = lane & 15;
    const size_t rowWG = (size_t)blockIdx.y * 128;
    const size_t colWG = (size_t)blockIdx.x * 128;

    // staging assignment: 256 threads x one 16B chunk per operand
    const int srow = tid >> 1;          // 0..127
    const int scol = (tid & 1) * 16;    // 0 / 16 (elements)
    const unsigned short* Ag = A  + (rowWG + srow) * (size_t)K + scol;
    const unsigned short* Bg = Bt + (colWG + srow) * (size_t)K + scol;
    const int soff = srow * 32 + scol;

    v8f acc[4][2];
#pragma unroll
    for (int i = 0; i < 4; ++i)
#pragma unroll
        for (int j = 0; j < 2; ++j)
#pragma unroll
            for (int e = 0; e < 8; ++e) acc[i][j][e] = 0.f;

    const int aoff = half ? 16 : 0;

    // prologue: stage k0 = 0 into buffer 0
    cp16_to_lds(Ag, &As[0][soff]);
    cp16_to_lds(Bg, &Bs[0][soff]);

    int cur = 0;
    for (int k0 = 0; k0 < K; k0 += 32, cur ^= 1) {
        async_publish_wait();
        __syncthreads();

        if (k0 + 32 < K) {
            cp16_to_lds(Ag + k0 + 32, &As[cur ^ 1][soff]);
            cp16_to_lds(Bg + k0 + 32, &Bs[cur ^ 1][soff]);
            if (k0 + 128 < K) {
                __builtin_prefetch((const void*)(Ag + k0 + 128), 0, 1);
                __builtin_prefetch((const void*)(Bg + k0 + 128), 0, 1);
            }
        }

        const unsigned short* Ab = As[cur];
        const unsigned short* Bb = Bs[cur];

        v16bf af[4], bf[2];
#pragma unroll
        for (int i = 0; i < 4; ++i) {
            const unsigned short* p = Ab + (wm * 64 + i * 16 + l15) * 32 + aoff;
            af[i] = frag_split(p, p + 8);
        }
#pragma unroll
        for (int j = 0; j < 2; ++j)
            bf[j] = frag_contig(Bb + (wn * 32 + j * 16 + l15) * 32 + half * 16);

#pragma unroll
        for (int i = 0; i < 4; ++i)
#pragma unroll
            for (int j = 0; j < 2; ++j)
                acc[i][j] = wmma_bf16(af[i], bf[j], acc[i][j]);
    }

#pragma unroll
    for (int i = 0; i < 4; ++i) {
#pragma unroll
        for (int j = 0; j < 2; ++j) {
            size_t col = colWG + (size_t)wn * 32 + (size_t)j * 16 + l15;
            float bv = ((EPI == 0 || EPI == 3) && bias) ? bias[col] : 0.f;
#pragma unroll
            for (int e = 0; e < 8; ++e) {
                size_t row = rowWG + (size_t)wm * 64 + (size_t)i * 16 + e + half * 8;
                size_t idx = row * (size_t)N + col;
                float v = acc[i][j][e];
                if (EPI == 0) {
                    outB[idx] = f2bf(v + bv);
                } else if (EPI == 1) {
                    outF[idx] += v;
                } else if (EPI == 2) {
                    float g = bf2f(aux[idx]);
                    float s = 1.f / (1.f + __expf(-g));
                    outB[idx] = f2bf(s * v);
                } else {
                    float x = v + bv;
                    outB[idx] = f2bf(0.5f * x * (1.f + erff(x * 0.70710678118f)));
                }
            }
        }
    }
}

// ============================================================================
// Flash attention (full, non-causal; mask over keys). One wave per 16-query
// tile. QK^T and PV both via WMMA; P & V^T staged through LDS.
// ============================================================================
__global__ __launch_bounds__(32)
void attn_fa(const unsigned short* __restrict__ q,   // [T, H]  bf16
             const unsigned short* __restrict__ k,   // [T, KVD] bf16 (roped)
             const unsigned short* __restrict__ v,   // [T, KVD] bf16
             const float* __restrict__ mask,         // [B, S]
             unsigned short* __restrict__ o)         // [T, H]  bf16
{
    __shared__ __align__(32) unsigned short Pt[16 * 32];   // P  tile (16q x 32k)
    __shared__ __align__(32) unsigned short Vt[64 * 32];   // V^T tile (64dh x 32k)

    const int lane = threadIdx.x;
    const int half = lane >> 4;
    const int l15  = lane & 15;
    const int q0   = blockIdx.x * 16;
    const int h    = blockIdx.y;
    const int b    = blockIdx.z;
    const int kvh  = h / (NHD / NKVH);
    const size_t bS = (size_t)b * SS;
    const float scale = 0.125f;   // 1/sqrt(64)

    // Q fragments (two K-steps over dh: 0..31, 32..63)
    v16bf qf[2];
    {
        const size_t base = (bS + q0 + l15) * (size_t)HH + (size_t)h * DHD;
        const int o0 = half ? 16 : 0;
        qf[0] = frag_split(q + base + o0,      q + base + o0 + 8);
        qf[1] = frag_split(q + base + 32 + o0, q + base + 32 + o0 + 8);
    }

    float m[8], lsum[8];
    v8f acc[4];
#pragma unroll
    for (int e = 0; e < 8; ++e) { m[e] = -1e30f; lsum[e] = 0.f; }
#pragma unroll
    for (int nt = 0; nt < 4; ++nt)
#pragma unroll
        for (int e = 0; e < 8; ++e) acc[nt][e] = 0.f;

    for (int kt = 0; kt < SS; kt += 32) {
        // ---- scores S = Q K^T (16q x 32k) : two N-tiles ----
        v8f s[2];
#pragma unroll
        for (int nt = 0; nt < 2; ++nt) {
            int key = kt + nt * 16 + l15;
            const size_t kb = (bS + key) * (size_t)KVD + (size_t)kvh * DHD + half * 16;
            v8f z;
#pragma unroll
            for (int e = 0; e < 8; ++e) z[e] = 0.f;
            z = wmma_bf16(qf[0], frag_contig(k + kb),      z);
            z = wmma_bf16(qf[1], frag_contig(k + kb + 32), z);
            s[nt] = z;
        }
        const float mk0 = mask[bS + kt + l15];
        const float mk1 = mask[bS + kt + 16 + l15];
#pragma unroll
        for (int e = 0; e < 8; ++e) {
            s[0][e] = s[0][e] * scale + (mk0 == 0.f ? -1e10f : 0.f);
            s[1][e] = s[1][e] * scale + (mk1 == 0.f ? -1e10f : 0.f);
        }

        // ---- online softmax (row stats live in each half-wave) ----
        float p0[8], p1[8];
#pragma unroll
        for (int e = 0; e < 8; ++e) {
            float mt = fmaxf(s[0][e], s[1][e]);
#pragma unroll
            for (int d = 1; d < 16; d <<= 1) mt = fmaxf(mt, __shfl_xor(mt, d, 32));
            float mnew  = fmaxf(m[e], mt);
            float alpha = __expf(m[e] - mnew);
            float a0 = __expf(s[0][e] - mnew);
            float a1 = __expf(s[1][e] - mnew);
            float rs = a0 + a1;
#pragma unroll
            for (int d = 1; d < 16; d <<= 1) rs += __shfl_xor(rs, d, 32);
            lsum[e] = lsum[e] * alpha + rs;
            m[e]    = mnew;
            p0[e] = a0; p1[e] = a1;
            acc[0][e] *= alpha; acc[1][e] *= alpha;
            acc[2][e] *= alpha; acc[3][e] *= alpha;
        }

        // ---- P (C-layout) -> LDS ----
#pragma unroll
        for (int e = 0; e < 8; ++e) {
            int M = e + half * 8;
            Pt[M * 32 + l15]      = f2bf(p0[e]);
            Pt[M * 32 + 16 + l15] = f2bf(p1[e]);
        }
        __syncthreads();

        // ---- stage V^T tile: Vt[dh][key] ----
        {
            const size_t vb = (bS + kt + lane) * (size_t)KVD + (size_t)kvh * DHD;
#pragma unroll
            for (int c = 0; c < 8; ++c) {
                u16x8 vv = *(const u16x8*)(v + vb + c * 8);
#pragma unroll
                for (int e = 0; e < 8; ++e) Vt[(c * 8 + e) * 32 + lane] = vv[e];
            }
        }
        __syncthreads();

        // ---- O += P V : A = P (16x32) from LDS, B = V^T rows from LDS ----
        v16bf pf;
        {
            const int o0 = half ? 16 : 0;
            pf = frag_split(Pt + l15 * 32 + o0, Pt + l15 * 32 + o0 + 8);
        }
#pragma unroll
        for (int nt = 0; nt < 4; ++nt) {
            const unsigned short* vp = Vt + (nt * 16 + l15) * 32 + half * 16;
            acc[nt] = wmma_bf16(pf, frag_contig(vp), acc[nt]);
        }
        __syncthreads();
    }

    // ---- finalize: O / l ----
#pragma unroll
    for (int e = 0; e < 8; ++e) {
        float inv = 1.f / lsum[e];
        size_t row = bS + q0 + e + half * 8;
#pragma unroll
        for (int nt = 0; nt < 4; ++nt)
            o[row * (size_t)HH + (size_t)h * DHD + nt * 16 + l15] =
                f2bf(acc[nt][e] * inv);
    }
}

// ============================================================================
// Elementwise / reduction kernels
// ============================================================================
__global__ void convT_k(const float* __restrict__ W,       // [K, N] fp32
                        unsigned short* __restrict__ Wt,   // [N, K] bf16
                        int K, int N)
{
    size_t idx = (size_t)blockIdx.x * blockDim.x + threadIdx.x;
    size_t total = (size_t)K * N;
    if (idx >= total) return;
    size_t n = idx / K, kk = idx % K;
    Wt[idx] = f2bf(W[kk * (size_t)N + n]);
}

__global__ void embed_k(const int* __restrict__ ids, const float* __restrict__ emb,
                        float* __restrict__ x)
{
    size_t idx = (size_t)blockIdx.x * blockDim.x + threadIdx.x;
    if (idx >= (size_t)TT * HH) return;
    size_t t = idx / HH; int c = (int)(idx % HH);
    x[idx] = emb[(size_t)ids[t] * HH + c];
}

__global__ __launch_bounds__(256)
void rmsnorm_k(const float* __restrict__ x, const float* __restrict__ w,
               unsigned short* __restrict__ hout)
{
    __shared__ float red[8];
    const int t = blockIdx.x;
    const float* xr = x + (size_t)t * HH;
    float ss = 0.f;
    for (int i = threadIdx.x; i < HH; i += 256) { float v = xr[i]; ss += v * v; }
#pragma unroll
    for (int d = 1; d < 32; d <<= 1) ss += __shfl_xor(ss, d, 32);
    if ((threadIdx.x & 31) == 0) red[threadIdx.x >> 5] = ss;
    __syncthreads();
    float tot = 0.f;
#pragma unroll
    for (int i = 0; i < 8; ++i) tot += red[i];
    float inv = rsqrtf(tot / (float)HH + 1e-6f);
    for (int i = threadIdx.x; i < HH; i += 256)
        hout[(size_t)t * HH + i] = f2bf(w[i] * xr[i] * inv);
}

__global__ void rope_k(unsigned short* __restrict__ buf,
                       const float* __restrict__ cosb, const float* __restrict__ sinb,
                       int nheads, int stride)
{
    size_t idx = (size_t)blockIdx.x * blockDim.x + threadIdx.x;
    size_t total = (size_t)TT * nheads * 32;
    if (idx >= total) return;
    int d = (int)(idx & 31);
    size_t tmp = idx >> 5;
    int hd = (int)(tmp % nheads);
    size_t t = tmp / nheads;
    int s = (int)(t % SS);
    size_t base = t * (size_t)stride + (size_t)hd * DHD;
    float x0 = bf2f(buf[base + d]);
    float x1 = bf2f(buf[base + 32 + d]);
    float c0 = cosb[s * DHD + d],      c1 = cosb[s * DHD + 32 + d];
    float s0 = sinb[s * DHD + d],      s1 = sinb[s * DHD + 32 + d];
    buf[base + d]      = f2bf(x0 * c0 - x1 * s0);
    buf[base + 32 + d] = f2bf(x1 * c1 + x0 * s1);
}

__global__ __launch_bounds__(256)
void head_k(const unsigned short* __restrict__ hg, const float* __restrict__ W2,
            const float* __restrict__ b2, float* __restrict__ out)
{
    __shared__ float red[8];
    const int t = blockIdx.x;
    float ss = 0.f;
    for (int i = threadIdx.x; i < HH; i += 256)
        ss += bf2f(hg[(size_t)t * HH + i]) * W2[i];
#pragma unroll
    for (int d = 1; d < 32; d <<= 1) ss += __shfl_xor(ss, d, 32);
    if ((threadIdx.x & 31) == 0) red[threadIdx.x >> 5] = ss;
    __syncthreads();
    if (threadIdx.x == 0) {
        float tot = 0.f;
#pragma unroll
        for (int i = 0; i < 8; ++i) tot += red[i];
        out[t] = tot + b2[0];
    }
}

// ============================================================================
// Orchestration
// ============================================================================
extern "C" void kernel_launch(void* const* d_in, const int* in_sizes, int n_in,
                              void* d_out, int out_size, void* d_ws, size_t ws_size,
                              hipStream_t stream)
{
    (void)in_sizes; (void)n_in; (void)out_size; (void)ws_size;

    const int*   ids   = (const int*)  d_in[0];
    const float* maskf = (const float*)d_in[1];
    const float* emb   = (const float*)d_in[2];
    const float* Wq    = (const float*)d_in[3];
    const float* bq    = (const float*)d_in[4];
    const float* Wk    = (const float*)d_in[5];
    const float* bk    = (const float*)d_in[6];
    const float* Wv    = (const float*)d_in[7];
    const float* bv    = (const float*)d_in[8];
    const float* Wo    = (const float*)d_in[9];
    const float* ln1   = (const float*)d_in[10];
    const float* ln2   = (const float*)d_in[11];
    const float* Wg    = (const float*)d_in[12];
    const float* Wu    = (const float*)d_in[13];
    const float* Wd    = (const float*)d_in[14];
    const float* normw = (const float*)d_in[15];
    const float* W1    = (const float*)d_in[16];
    const float* b1    = (const float*)d_in[17];
    const float* W2    = (const float*)d_in[18];
    const float* b2    = (const float*)d_in[19];
    const float* cosb  = (const float*)d_in[20];
    const float* sinb  = (const float*)d_in[21];

    char* ws = (char*)d_ws;
    size_t off = 0;
    auto carve = [&](size_t bytes) -> char* {
        size_t p = (off + 255) & ~(size_t)255;
        off = p + bytes;
        return ws + p;
    };

    // bf16 transposed weights ([N,K] layout)
    unsigned short *wq_t[LLAY], *wk_t[LLAY], *wv_t[LLAY], *wo_t[LLAY];
    unsigned short *wg_t[LLAY], *wu_t[LLAY], *wd_t[LLAY];
    for (int l = 0; l < LLAY; ++l) {
        wq_t[l] = (unsigned short*)carve((size_t)HH * HH  * 2);
        wk_t[l] = (unsigned short*)carve((size_t)KVD * HH * 2);
        wv_t[l] = (unsigned short*)carve((size_t)KVD * HH * 2);
        wo_t[l] = (unsigned short*)carve((size_t)HH * HH  * 2);
        wg_t[l] = (unsigned short*)carve((size_t)MFF * HH * 2);
        wu_t[l] = (unsigned short*)carve((size_t)MFF * HH * 2);
        wd_t[l] = (unsigned short*)carve((size_t)HH * MFF * 2);
    }
    unsigned short* w1_t = (unsigned short*)carve((size_t)HH * HH * 2);

    // activations
    float*          xbuf = (float*)         carve((size_t)TT * HH  * 4);
    unsigned short* hbuf = (unsigned short*)carve((size_t)TT * HH  * 2);
    unsigned short* qb   = (unsigned short*)carve((size_t)TT * HH  * 2);
    unsigned short* kb   = (unsigned short*)carve((size_t)TT * KVD * 2);
    unsigned short* vb   = (unsigned short*)carve((size_t)TT * KVD * 2);
    unsigned short* ab   = (unsigned short*)carve((size_t)TT * HH  * 2);
    unsigned short* gb   = (unsigned short*)carve((size_t)TT * MFF * 2);
    unsigned short* ub   = (unsigned short*)carve((size_t)TT * MFF * 2);
    unsigned short* hg   = (unsigned short*)carve((size_t)TT * HH  * 2);

    auto cgrid = [](size_t n) { return dim3((unsigned)((n + 255) / 256)); };

    // ---- weight conversion (fp32 [K,N] -> bf16 [N,K]) ----
    for (int l = 0; l < LLAY; ++l) {
        convT_k<<<cgrid((size_t)HH * HH), 256, 0, stream>>>(Wq + (size_t)l * HH * HH,  wq_t[l], HH, HH);
        convT_k<<<cgrid((size_t)HH * KVD), 256, 0, stream>>>(Wk + (size_t)l * HH * KVD, wk_t[l], HH, KVD);
        convT_k<<<cgrid((size_t)HH * KVD), 256, 0, stream>>>(Wv + (size_t)l * HH * KVD, wv_t[l], HH, KVD);
        convT_k<<<cgrid((size_t)HH * HH), 256, 0, stream>>>(Wo + (size_t)l * HH * HH,  wo_t[l], HH, HH);
        convT_k<<<cgrid((size_t)HH * MFF), 256, 0, stream>>>(Wg + (size_t)l * HH * MFF, wg_t[l], HH, MFF);
        convT_k<<<cgrid((size_t)HH * MFF), 256, 0, stream>>>(Wu + (size_t)l * HH * MFF, wu_t[l], HH, MFF);
        convT_k<<<cgrid((size_t)MFF * HH), 256, 0, stream>>>(Wd + (size_t)l * MFF * HH, wd_t[l], MFF, HH);
    }
    convT_k<<<cgrid((size_t)HH * HH), 256, 0, stream>>>(W1, w1_t, HH, HH);

    // ---- embedding gather ----
    embed_k<<<cgrid((size_t)TT * HH), 256, 0, stream>>>(ids, emb, xbuf);

    const dim3 gH (HH  / 128, TT / 128);   // (12, 32)
    const dim3 gKV(KVD / 128, TT / 128);   // (2, 32)
    const dim3 gM (MFF / 128, TT / 128);   // (70, 32)

    for (int l = 0; l < LLAY; ++l) {
        // attention block
        rmsnorm_k<<<TT, 256, 0, stream>>>(xbuf, ln1 + (size_t)l * HH, hbuf);
        gemm_bf16<0><<<gH,  256, 0, stream>>>(hbuf, wq_t[l], bq + (size_t)l * HH,  nullptr, nullptr, qb, TT, HH,  HH);
        gemm_bf16<0><<<gKV, 256, 0, stream>>>(hbuf, wk_t[l], bk + (size_t)l * KVD, nullptr, nullptr, kb, TT, KVD, HH);
        gemm_bf16<0><<<gKV, 256, 0, stream>>>(hbuf, wv_t[l], bv + (size_t)l * KVD, nullptr, nullptr, vb, TT, KVD, HH);
        rope_k<<<cgrid((size_t)TT * NHD  * 32), 256, 0, stream>>>(qb, cosb, sinb, NHD,  HH);
        rope_k<<<cgrid((size_t)TT * NKVH * 32), 256, 0, stream>>>(kb, cosb, sinb, NKVH, KVD);
        attn_fa<<<dim3(SS / 16, NHD, BB), 32, 0, stream>>>(qb, kb, vb, maskf, ab);
        gemm_bf16<1><<<gH, 256, 0, stream>>>(ab, wo_t[l], nullptr, nullptr, xbuf, nullptr, TT, HH, HH);
        // MLP block (SwiGLU)
        rmsnorm_k<<<TT, 256, 0, stream>>>(xbuf, ln2 + (size_t)l * HH, hbuf);
        gemm_bf16<0><<<gM, 256, 0, stream>>>(hbuf, wg_t[l], nullptr, nullptr, nullptr, gb, TT, MFF, HH);
        gemm_bf16<2><<<gM, 256, 0, stream>>>(hbuf, wu_t[l], nullptr, gb,      nullptr, ub, TT, MFF, HH);
        gemm_bf16<1><<<gH, 256, 0, stream>>>(ub,   wd_t[l], nullptr, nullptr, xbuf, nullptr, TT, HH, MFF);
    }

    // ---- head ----
    rmsnorm_k<<<TT, 256, 0, stream>>>(xbuf, normw, hbuf);
    gemm_bf16<3><<<gH, 256, 0, stream>>>(hbuf, w1_t, b1, nullptr, nullptr, hg, TT, HH, HH);
    head_k<<<TT, 256, 0, stream>>>(hg, W2, b2, (float*)d_out);
}
